// MixtureOfExperts_4398046511756
// MI455X (gfx1250) — matrix operand order repaired and checked by reading, not compiled, not run on hardware
//
#include <hip/hip_runtime.h>
#include <math.h>

typedef __bf16 bf16;
typedef __attribute__((ext_vector_type(16))) __bf16 v16bf;
typedef __attribute__((ext_vector_type(8)))  __bf16 v8bf;
typedef __attribute__((ext_vector_type(8)))  float  v8f;

#define D_MODEL 1024
#define HIDDEN  2048
#define NEXP    8
#define TOKENS  8192
#define A_STRIDE 1032   // 1024 + 8 bf16 pad: 16B aligned, spreads LDS banks
#define H_STRIDE 264    // 256 + 8 bf16 pad

union AFrag { v16bf v; v8bf h[2]; };
union CFrag { v8f v; float f[8]; };

__device__ __forceinline__ float gelu_exact(float x) {
  return 0.5f * x * (1.0f + erff(x * 0.70710678118654752f));
}

__device__ __forceinline__ float wave_sum(float v) {
#pragma unroll
  for (int m = 16; m >= 1; m >>= 1) v += __shfl_xor(v, m, 32);
  return v;
}

// ---------------------------------------------------------------------------
// Kernel 1: fp32 -> bf16 weight conversion into workspace + zero counters.
// ---------------------------------------------------------------------------
__global__ void moe_prep(const float* __restrict__ w1, const float* __restrict__ w2,
                         bf16* __restrict__ w1b, bf16* __restrict__ w2b,
                         int* __restrict__ cnt) {
  size_t i = ((size_t)blockIdx.x * blockDim.x + threadIdx.x) * 4;
  float4 a = *(const float4*)(w1 + i);
  w1b[i + 0] = (bf16)a.x; w1b[i + 1] = (bf16)a.y;
  w1b[i + 2] = (bf16)a.z; w1b[i + 3] = (bf16)a.w;
  float4 b = *(const float4*)(w2 + i);
  w2b[i + 0] = (bf16)b.x; w2b[i + 1] = (bf16)b.y;
  w2b[i + 2] = (bf16)b.z; w2b[i + 3] = (bf16)b.w;
  if (blockIdx.x == 0 && threadIdx.x < NEXP) cnt[threadIdx.x] = 0;
}

// ---------------------------------------------------------------------------
// Kernel 2: RMSNorm + gate top-2 routing + residual init. One wave per token.
// ---------------------------------------------------------------------------
__global__ void moe_route(const float* __restrict__ x, const float* __restrict__ nw,
                          const float* __restrict__ gw, bf16* __restrict__ tokb,
                          float* __restrict__ out, int* __restrict__ cnt,
                          int* __restrict__ etok, float* __restrict__ ewgt) {
  const int t    = blockIdx.x * 8 + (threadIdx.x >> 5);
  const int lane = threadIdx.x & 31;
  const size_t base = (size_t)t * D_MODEL + (size_t)lane * 32;

  float4 vx[8], nv[8];
  const float4* xp = (const float4*)(x + base);
  float ss = 0.f;
#pragma unroll
  for (int i = 0; i < 8; ++i) {
    vx[i] = xp[i];
    ss += vx[i].x * vx[i].x + vx[i].y * vx[i].y +
          vx[i].z * vx[i].z + vx[i].w * vx[i].w;
  }
  ss = wave_sum(ss);
  const float r = rsqrtf(ss * (1.0f / D_MODEL) + 1e-6f);

  const float4* np = (const float4*)(nw + (size_t)lane * 32);
#pragma unroll
  for (int i = 0; i < 8; ++i) {
    float4 w4 = np[i];
    nv[i].x = vx[i].x * r * w4.x;
    nv[i].y = vx[i].y * r * w4.y;
    nv[i].z = vx[i].z * r * w4.z;
    nv[i].w = vx[i].w * r * w4.w;
  }

  // residual init: out = x
  float4* op = (float4*)(out + base);
#pragma unroll
  for (int i = 0; i < 8; ++i) op[i] = vx[i];

  // normalized token in bf16 for the WMMA GEMMs
  bf16* tp = tokb + base;
#pragma unroll
  for (int i = 0; i < 8; ++i) {
    tp[i * 4 + 0] = (bf16)nv[i].x;
    tp[i * 4 + 1] = (bf16)nv[i].y;
    tp[i * 4 + 2] = (bf16)nv[i].z;
    tp[i * 4 + 3] = (bf16)nv[i].w;
  }

  // gate logits: 8 dot products of length 1024
  float ga[NEXP];
#pragma unroll
  for (int e = 0; e < NEXP; ++e) {
    const float4* gp = (const float4*)(gw + (size_t)e * D_MODEL + (size_t)lane * 32);
    float acc = 0.f;
#pragma unroll
    for (int i = 0; i < 8; ++i) {
      float4 g4 = gp[i];
      acc += g4.x * nv[i].x + g4.y * nv[i].y + g4.z * nv[i].z + g4.w * nv[i].w;
    }
    ga[e] = wave_sum(acc);
  }

  if (lane == 0) {
    int i0 = 0; float g0 = ga[0];
#pragma unroll
    for (int e2 = 1; e2 < NEXP; ++e2) if (ga[e2] > g0) { g0 = ga[e2]; i0 = e2; }
    int i1 = 0; float g1 = -3.4e38f;
#pragma unroll
    for (int e2 = 0; e2 < NEXP; ++e2)
      if (e2 != i0 && ga[e2] > g1) { g1 = ga[e2]; i1 = e2; }
    const float e1 = expf(g1 - g0);          // g0 >= g1
    const float s  = 1.0f + e1;
    const float wt0 = 1.0f / s, wt1 = e1 / s;
    int p0 = atomicAdd(&cnt[i0], 1);
    etok[i0 * TOKENS + p0] = t; ewgt[i0 * TOKENS + p0] = wt0;
    int p1 = atomicAdd(&cnt[i1], 1);
    etok[i1 * TOKENS + p1] = t; ewgt[i1 * TOKENS + p1] = wt1;
  }
}

// ---------------------------------------------------------------------------
// Kernel 3: routed expert FFN. Block = 8 waves = one (expert, 16-token) tile.
// Fused GEMM1 -> GELU -> GEMM2 with bf16 WMMA, fp32 accumulation.
// ---------------------------------------------------------------------------
__global__ void __launch_bounds__(256) moe_ffn(
    const bf16* __restrict__ tokb, const bf16* __restrict__ w1b,
    const bf16* __restrict__ w2b, const float* __restrict__ b1g,
    const float* __restrict__ b2g, const int* __restrict__ cnt,
    const int* __restrict__ etok, const float* __restrict__ ewgt,
    float* __restrict__ out) {
  const int e    = blockIdx.y;
  const int tile = blockIdx.x;
  const int count = cnt[e];
  if (tile * 16 >= count) return;

  __shared__ __align__(32) bf16 A_lds[16 * A_STRIDE];   // 16 x 1024 activations
  __shared__ __align__(32) bf16 h_lds[16 * H_STRIDE];   // 16 x 256 hidden chunk
  __shared__ int   tok_s[16];
  __shared__ float wgt_s[16];

  const int tid  = threadIdx.x;
  const int w    = tid >> 5;     // wave id 0..7
  const int lane = tid & 31;
  const int m16  = lane & 15;
  const int half = lane >> 4;

  if (tid < 16) {
    const int row = tile * 16 + tid;
    const int src = e * TOKENS + ((row < count) ? row : tile * 16);
    tok_s[tid] = etok[src];
    wgt_s[tid] = (row < count) ? ewgt[src] : 0.0f;   // padded rows contribute 0
  }
  __syncthreads();

  // Stage A tile (gathered tokens) into LDS, 16B vector copies.
  for (int v = tid; v < 16 * (D_MODEL / 8); v += 256) {
    const int rr = v >> 7;
    const int cc = (v & 127) * 8;
    *(v8bf*)&A_lds[rr * A_STRIDE + cc] =
        *(const v8bf*)(tokb + (size_t)tok_s[rr] * D_MODEL + cc);
  }
  __syncthreads();

  const bf16* w1e = w1b + (size_t)e * HIDDEN * D_MODEL;
  const bf16* w2e = w2b + (size_t)e * D_MODEL * HIDDEN;

  CFrag acc[8] = {};   // wave owns out columns [w*128, w*128+128): 8 subtiles

  for (int hc = 0; hc < HIDDEN / 256; ++hc) {
    const int hcb = hc * 256;

    // ---- GEMM1: h[16, 32-per-wave] = A[16,1024] @ W1^T, K-loop over D ----
    CFrag c0 = {}, c1 = {};
    const int nA = hcb + w * 32 + m16;     // hidden column this lane owns
    const int nB = nA + 16;
    const bf16* w1rA = w1e + (size_t)nA * D_MODEL + half * 16;
    const bf16* w1rB = w1e + (size_t)nB * D_MODEL + half * 16;
#pragma unroll 4
    for (int kk = 0; kk < D_MODEL / 32; ++kk) {
      const int kb = kk * 32;
      AFrag a;     // A 16x32 bf16 frag: two 16B runs per lane
      a.h[0] = *(const v8bf*)&A_lds[m16 * A_STRIDE + kb + half * 16];
      a.h[1] = *(const v8bf*)&A_lds[m16 * A_STRIDE + kb + half * 16 + 8];
      v16bf b0 = *(const v16bf*)(w1rA + kb);   // B 32x16: 16 contiguous K
      v16bf b1 = *(const v16bf*)(w1rB + kb);
      c0.v = __builtin_amdgcn_wmma_f32_16x16x32_bf16(false, a.v, false, b0,
                                                     (short)0, c0.v, false, false);
      c1.v = __builtin_amdgcn_wmma_f32_16x16x32_bf16(false, a.v, false, b1,
                                                     (short)0, c1.v, false, false);
    }

    __syncthreads();   // all waves done reading h_lds from previous chunk
    {
      const float bvA = b1g[(size_t)e * HIDDEN + nA];
      const float bvB = b1g[(size_t)e * HIDDEN + nB];
      const int colA = w * 32 + m16;
#pragma unroll
      for (int rrow = 0; rrow < 8; ++rrow) {
        const int mrow = rrow + half * 8;
        h_lds[mrow * H_STRIDE + colA]      = (bf16)gelu_exact(c0.f[rrow] + bvA);
        h_lds[mrow * H_STRIDE + colA + 16] = (bf16)gelu_exact(c1.f[rrow] + bvB);
      }
    }
    __syncthreads();

    // ---- GEMM2: acc[16,1024] += h_chunk[16,256] @ W2^T chunk ----
#pragma unroll
    for (int ks = 0; ks < 8; ++ks) {
      const int kb = ks * 32;
      AFrag hf;
      hf.h[0] = *(const v8bf*)&h_lds[m16 * H_STRIDE + kb + half * 16];
      hf.h[1] = *(const v8bf*)&h_lds[m16 * H_STRIDE + kb + half * 16 + 8];
      const int kg = hcb + kb + half * 16;
#pragma unroll
      for (int s = 0; s < 8; ++s) {
        const int n = w * 128 + s * 16 + m16;
        v16bf bb = *(const v16bf*)(w2e + (size_t)n * HIDDEN + kg);
        acc[s].v = __builtin_amdgcn_wmma_f32_16x16x32_bf16(false, hf.v, false, bb,
                                                           (short)0, acc[s].v,
                                                           false, false);
      }
    }
  }

  // Epilogue: out[token] += weight * (acc + b2)
#pragma unroll
  for (int s = 0; s < 8; ++s) {
    const int n = w * 128 + s * 16 + m16;
    const float b2v = b2g[(size_t)e * D_MODEL + n];
#pragma unroll
    for (int rrow = 0; rrow < 8; ++rrow) {
      const int mrow = rrow + half * 8;
      const float val = (acc[s].f[rrow] + b2v) * wgt_s[mrow];
      atomicAdd(&out[(size_t)tok_s[mrow] * D_MODEL + n], val);
    }
  }
}

// ---------------------------------------------------------------------------
extern "C" void kernel_launch(void* const* d_in, const int* in_sizes, int n_in,
                              void* d_out, int out_size, void* d_ws, size_t ws_size,
                              hipStream_t stream) {
  const float* x  = (const float*)d_in[0];
  const float* nw = (const float*)d_in[1];
  const float* gw = (const float*)d_in[2];
  const float* w1 = (const float*)d_in[3];
  const float* b1 = (const float*)d_in[4];
  const float* w2 = (const float*)d_in[5];
  const float* b2 = (const float*)d_in[6];
  float* out = (float*)d_out;

  // Workspace layout (all 256B aligned):
  //   tokens bf16   : 8192*1024*2        = 16 MB
  //   w1 bf16       : 8*2048*1024*2      = 32 MB
  //   w2 bf16       : 8*1024*2048*2      = 32 MB
  //   counters      : 8*4 (padded 256B)
  //   expert tokens : 8*8192*4           = 256 KB
  //   expert weights: 8*8192*4           = 256 KB
  char* ws = (char*)d_ws;
  const size_t off_tok  = 0;
  const size_t off_w1b  = off_tok + (size_t)TOKENS * D_MODEL * 2;
  const size_t off_w2b  = off_w1b + (size_t)NEXP * HIDDEN * D_MODEL * 2;
  const size_t off_cnt  = off_w2b + (size_t)NEXP * D_MODEL * HIDDEN * 2;
  const size_t off_etok = off_cnt + 256;
  const size_t off_ewgt = off_etok + (size_t)NEXP * TOKENS * 4;

  bf16*  tokb = (bf16*)(ws + off_tok);
  bf16*  w1bp = (bf16*)(ws + off_w1b);
  bf16*  w2bp = (bf16*)(ws + off_w2b);
  int*   cnt  = (int*)(ws + off_cnt);
  int*   etok = (int*)(ws + off_etok);
  float* ewgt = (float*)(ws + off_ewgt);

  // 1) weight conversion + counter reset (16,777,216 elems / 4 per thread)
  moe_prep<<<16384, 256, 0, stream>>>(w1, w2, w1bp, w2bp, cnt);
  // 2) rmsnorm + gate + routing + residual (one wave per token)
  moe_route<<<TOKENS / 8, 256, 0, stream>>>(x, nw, gw, tokb, out, cnt, etok, ewgt);
  // 3) routed expert FFN (early-exits past each expert's token count)
  dim3 grid(TOKENS / 16, NEXP);
  moe_ffn<<<grid, 256, 0, stream>>>(tokb, w1bp, w2bp, b1, b2, cnt, etok, ewgt, out);
}
